// GeneralizedActionRNNGCell_44083544326935
// MI455X (gfx1250) — compile-verified, matrix operand order checked by reading, not run on hardware
//
#include <hip/hip_runtime.h>
#include <hip/hip_bf16.h>

// ---------------- problem constants ----------------
constexpr int B  = 2048;
constexpr int L  = 32;
constexpr int H  = 512;
constexpr int H2 = 2 * H;   // 1024
constexpr int H4 = 4 * H;   // 2048

typedef _Float16 h16;
typedef __attribute__((ext_vector_type(16))) _Float16 v16h;
typedef __attribute__((ext_vector_type(8)))  float    v8f;

union V16H { v16h v; uint4 u[2]; };

// Load one 16x32 (f16) WMMA operand fragment for this lane (global or LDS source).
// ISA 7.12.2: lanes 0-15 hold K=[0..7]+[16..23], lanes 16-31 hold K=[8..15]+[24..31].
__device__ inline v16h load_frag(const h16* __restrict__ rowptr, int k0, int khalf) {
  V16H r;
  r.u[0] = *reinterpret_cast<const uint4*>(rowptr + k0 + khalf);
  r.u[1] = *reinterpret_cast<const uint4*>(rowptr + k0 + khalf + 16);
  return r.v;
}

__device__ inline float sigf(float x) { return 1.0f / (1.0f + __expf(-x)); }

// Async copy 16B global -> LDS (ASYNCcnt-tracked, CDNA5).
__device__ inline void async_g2l_b128(uint32_t lds_byte_addr, const void* gptr) {
  asm volatile("global_load_async_to_lds_b128 %0, %1, off"
               :: "v"(lds_byte_addr), "v"(gptr) : "memory");
}
__device__ inline void wait_asynccnt0() {
  asm volatile("s_wait_asynccnt 0x0" ::: "memory");
}

// ---------------- elementwise helpers ----------------
__global__ void k_f32_to_h16(const float* __restrict__ x, h16* __restrict__ y, size_t n) {
  size_t i = (size_t)blockIdx.x * blockDim.x + threadIdx.x;
  if (i < n) y[i] = (h16)x[i];
}

__global__ void k_zero_u32(unsigned* __restrict__ p, size_t n) {
  size_t i = (size_t)blockIdx.x * blockDim.x + threadIdx.x;
  if (i < n) p[i] = 0u;
}

__global__ void k_add2(const float* __restrict__ a, const float* __restrict__ b,
                       float* __restrict__ o, int n) {
  int i = blockIdx.x * blockDim.x + threadIdx.x;
  if (i < n) o[i] = a[i] + b[i];
}

// Vt[e][d] = V[d][e], output f16 (so the generic GEMM B-loader sees [N][K])
__global__ void k_transpose_h16(const float* __restrict__ V, h16* __restrict__ Vt) {
  int i = blockIdx.x * blockDim.x + threadIdx.x;   // over H2*H2
  if (i < H2 * H2) {
    int d = i & (H2 - 1);
    int e = i >> 10;
    Vt[(size_t)e * H2 + d] = (h16)V[(size_t)d * H2 + e];
  }
}

// rhs[b] = concat(nt_emb[nt_id[b]], stack_state[b]) in f16
__global__ void k_build_rhs(const float* __restrict__ nt_emb, const int* __restrict__ nt_id,
                            const float* __restrict__ stack, h16* __restrict__ rhs) {
  int i = blockIdx.x * blockDim.x + threadIdx.x;   // over B*H2
  if (i < B * H2) {
    int e = i & (H2 - 1);
    int b = i >> 10;
    float v = (e < H) ? nt_emb[(size_t)nt_id[b] * H + e]
                      : stack[(size_t)b * H + (e - H)];
    rhs[i] = (h16)v;
  }
}

// ---------------- generic WMMA GEMM: C[M,N] = A[M,K] * Bw[N,K]^T (+bias)(+act) ----------------
// one wave per block; each wave computes a 64x64 tile (4x4 sub-tiles of 16x16x32)
__global__ __launch_bounds__(32)
void k_wmma_gemm64(const h16* __restrict__ A, int lda,
                   const h16* __restrict__ Bw, int ldb,
                   const float* __restrict__ bias,
                   float* __restrict__ Cf, int ldc,
                   int K, int act) {
  const int lane  = threadIdx.x & 31;
  const int khalf = (lane >> 4) * 8;
  const int mrow  = lane & 15;
  const int n0 = blockIdx.x * 64;
  const int m0 = blockIdx.y * 64;

  const h16* arow[4];
  const h16* brow[4];
#pragma unroll
  for (int t = 0; t < 4; ++t) {
    arow[t] = A + (size_t)(m0 + t * 16 + mrow) * lda;
    brow[t] = Bw + (size_t)(n0 + t * 16 + mrow) * ldb;
  }

  v8f acc[4][4] = {};
  for (int k = 0; k < K; k += 32) {
    v16h af[4], bf[4];
#pragma unroll
    for (int t = 0; t < 4; ++t) af[t] = load_frag(arow[t], k, khalf);
#pragma unroll
    for (int t = 0; t < 4; ++t) bf[t] = load_frag(brow[t], k, khalf);
#pragma unroll
    for (int i = 0; i < 4; ++i)
#pragma unroll
      for (int j = 0; j < 4; ++j)
        acc[i][j] = __builtin_amdgcn_wmma_f32_16x16x32_f16(
            false, af[i], false, bf[j], (short)0, acc[i][j], false, false);
  }

  const int cn = lane & 15;
  const int mbase = (lane >> 4) * 8;
#pragma unroll
  for (int i = 0; i < 4; ++i)
#pragma unroll
    for (int j = 0; j < 4; ++j) {
      int col = n0 + j * 16 + cn;
      float bv = bias ? bias[col] : 0.0f;
#pragma unroll
      for (int r = 0; r < 8; ++r) {
        float v = acc[i][j][r] + bv;
        if (act == 1) v = fmaxf(v, 0.0f);
        int row = m0 + i * 16 + mbase + r;
        Cf[(size_t)row * ldc + col] = v;
      }
    }
}

// ---------------- fused LSTM step: GEMM (x*Wih^T + h*Whh^T) + cell update ----------------
// Tiling: block = 128 threads (4 waves). Wave w owns batch rows [m0+w*64, +64).
// All waves share the same 16 hidden columns nh0..nh0+15; a wave's 4 N-subtiles are the
// FOUR GATE BLOCKS (cols g*512 + nh0 + n), so i/f/g/o of each (b,n) land in one lane's
// accumulators and the cell nonlinearity fuses into the epilogue (no gates buffer).
// Weight fragments (64 rows x 32 K) are staged in LDS via global_load_async_to_lds_b128
// (ASYNCcnt) and consumed with ds_load_b128; A-operand rows are per-lane direct global
// (the bwd direction needs the per-row packed-sequence reversal anyway).
// grid: (H/16, B/256, 2 dirs)
__global__ __launch_bounds__(128)
void k_lstm_step_fused(const h16* __restrict__ ch, const int* __restrict__ len,
                       const h16* __restrict__ h_prev, h16* __restrict__ h_next,
                       float* __restrict__ c_state,
                       const h16* __restrict__ Wih_f, const h16* __restrict__ Whh_f,
                       const h16* __restrict__ Wih_b, const h16* __restrict__ Whh_b,
                       const float* __restrict__ bias_f, const float* __restrict__ bias_b,
                       const h16* __restrict__ zerovec,
                       h16* __restrict__ h_all, int l) {
  __shared__ h16 sB[64 * 32];   // 64 weight rows x 32 K halves (4 KB)

  const int dir  = blockIdx.z;
  const int tid  = threadIdx.x;            // 0..127
  const int wave = tid >> 5;               // 0..3
  const int lane = tid & 31;
  const int khalf = (lane >> 4) * 8;
  const int mrow  = lane & 15;
  const int nh0 = blockIdx.x * 16;         // hidden-column tile
  const int m0  = blockIdx.y * 256 + wave * 64;

  const h16* Wih = dir ? Wih_b : Wih_f;
  const h16* Whh = dir ? Whh_b : Whh_f;
  const float* bias = dir ? bias_b : bias_f;

  const uint32_t sB_base = (uint32_t)(uintptr_t)(void*)sB;  // LDS byte address (low 32 bits)

  // per-lane A row pointers (x part handles fwd / packed-reverse bwd)
  const h16* xrow[4];
  const h16* hrow[4];
#pragma unroll
  for (int t = 0; t < 4; ++t) {
    int b = m0 + t * 16 + mrow;
    if (dir == 0) {
      xrow[t] = ch + ((size_t)b * L + l) * H;
    } else {
      int ln = len[b];
      xrow[t] = (l < ln) ? ch + ((size_t)b * L + (ln - 1 - l)) * H : zerovec;
    }
    hrow[t] = h_prev + ((size_t)dir * B + b) * H;
  }

  // staging map: weight row for sB row rb (rb = g*16 + nrow) at current k
  // each thread moves two b128 chunks per K-step
  v8f acc[4][4] = {};   // [i = M-subtile][g = gate]

#pragma unroll 1
  for (int phase = 0; phase < 2; ++phase) {
    const h16* Wsel = phase ? Whh : Wih;
#pragma unroll 1
    for (int k = 0; k < H; k += 32) {
      // ---- async stage B tile (64 rows x 32 halves) into LDS ----
#pragma unroll
      for (int q = 0; q < 2; ++q) {
        int idx = tid + q * 128;                 // 0..255 chunk id
        int rb = idx >> 2, c = idx & 3;          // row-in-tile, 8-half chunk
        int n = ((rb >> 4) << 9) + nh0 + (rb & 15);   // g*512 + nh0 + nrow
        const h16* src = Wsel + (size_t)n * H + k + c * 8;
        async_g2l_b128(sB_base + (uint32_t)(rb * 32 + c * 8) * 2, src);
      }
      wait_asynccnt0();
      __syncthreads();

      // ---- fragments + 16 WMMAs ----
      const h16* arow_sel[4];
#pragma unroll
      for (int t = 0; t < 4; ++t) arow_sel[t] = phase ? hrow[t] : xrow[t];
      v16h af[4], bf[4];
#pragma unroll
      for (int t = 0; t < 4; ++t) af[t] = load_frag(arow_sel[t], k, khalf);
#pragma unroll
      for (int g = 0; g < 4; ++g) bf[g] = load_frag(sB + (g * 16 + mrow) * 32, 0, khalf);
#pragma unroll
      for (int i = 0; i < 4; ++i)
#pragma unroll
        for (int g = 0; g < 4; ++g)
          acc[i][g] = __builtin_amdgcn_wmma_f32_16x16x32_f16(
              false, af[i], false, bf[g], (short)0, acc[i][g], false, false);
      __syncthreads();   // protect sB before next stage
    }
  }

  // ---- fused cell epilogue: i/f/g/o for (row, nh) are acc[i][0..3][r] ----
  const int cn = lane & 15;
  const int mbase = (lane >> 4) * 8;
  const int nh = nh0 + cn;
  const float bi = bias[nh];
  const float bfg = bias[H + nh];
  const float bg = bias[2 * H + nh];
  const float bo = bias[3 * H + nh];
#pragma unroll
  for (int i = 0; i < 4; ++i) {
#pragma unroll
    for (int r = 0; r < 8; ++r) {
      int row = m0 + i * 16 + mbase + r;
      size_t idx = ((size_t)dir * B + row) * H + nh;
      float gi = acc[i][0][r] + bi;
      float gf = acc[i][1][r] + bfg;
      float gg = acc[i][2][r] + bg;
      float go = acc[i][3][r] + bo;
      float c = c_state[idx];
      c = sigf(gf) * c + sigf(gi) * tanhf(gg);
      float hv = sigf(go) * tanhf(c);
      c_state[idx] = c;
      h_next[idx] = (h16)hv;
      int ln = len[row];
      if (l < ln) {
        if (dir == 0)
          h_all[((size_t)row * L + l) * H2 + nh] = (h16)hv;
        else
          h_all[((size_t)row * L + (ln - 1 - l)) * H2 + H + nh] = (h16)hv;
      }
    }
  }
}

// ---------------- fused attention + both layernorms ----------------
__device__ inline float block_sum256(float v, float* s, int tid) {
  s[tid] = v;
  __syncthreads();
  for (int off = 128; off > 0; off >>= 1) {
    if (tid < off) s[tid] += s[tid + off];
    __syncthreads();
  }
  float r = s[0];
  __syncthreads();
  return r;
}

__global__ __launch_bounds__(256)
void k_attn_fused(const h16* __restrict__ h_all, const float* __restrict__ u,
                  const int* __restrict__ len,
                  const float* __restrict__ nt_emb2, const int* __restrict__ nt_id,
                  const float* __restrict__ ln1_g, const float* __restrict__ ln1_b,
                  const float* __restrict__ ln2_g, const float* __restrict__ ln2_b,
                  float* __restrict__ attn_out, h16* __restrict__ gate_in,
                  float* __restrict__ nt2_f, float* __restrict__ wc_f) {
  const int b = blockIdx.x;
  const int tid = threadIdx.x;
  __shared__ float s_u[H2];
  __shared__ float s_part[256];
  __shared__ float s_logit[L];
  __shared__ float s_attn[L];

  for (int e = tid; e < H2; e += 256) s_u[e] = u[(size_t)b * H2 + e];
  __syncthreads();

  {
    int l = tid >> 3, sub = tid & 7;
    const h16* hrow = h_all + ((size_t)b * L + l) * H2;
    float s = 0.0f;
    int e0 = sub * 128;
    for (int e = e0; e < e0 + 128; ++e) s += (float)hrow[e] * s_u[e];
    s_part[tid] = s;
  }
  __syncthreads();
  if (tid < L) {
    float t = 0.0f;
#pragma unroll
    for (int j = 0; j < 8; ++j) t += s_part[tid * 8 + j];
    s_logit[tid] = (tid < len[b]) ? t : -3.0e38f;
  }
  __syncthreads();
  if (tid == 0) {
    float mx = -3.0e38f;
    for (int l = 0; l < L; ++l) mx = fmaxf(mx, s_logit[l]);
    float sm = 0.0f;
    for (int l = 0; l < L; ++l) { float e = __expf(s_logit[l] - mx); s_attn[l] = e; sm += e; }
    float inv = 1.0f / sm;
    for (int l = 0; l < L; ++l) {
      s_attn[l] *= inv;
      attn_out[(size_t)b * L + l] = s_attn[l];
    }
  }
  __syncthreads();

  float wc[4];
#pragma unroll
  for (int q = 0; q < 4; ++q) {
    int e = tid + q * 256;
    float a = 0.0f;
    for (int l = 0; l < L; ++l)
      a += s_attn[l] * (float)h_all[((size_t)b * L + l) * H2 + e];
    wc[q] = a;
  }
  {
    float su = wc[0] + wc[1] + wc[2] + wc[3];
    float mean = block_sum256(su, s_part, tid) * (1.0f / H2);
    float sq = 0.0f;
#pragma unroll
    for (int q = 0; q < 4; ++q) { float d = wc[q] - mean; sq += d * d; }
    float var = block_sum256(sq, s_part, tid) * (1.0f / H2);
    float rstd = rsqrtf(var + 1e-5f);
#pragma unroll
    for (int q = 0; q < 4; ++q) {
      int e = tid + q * 256;
      float v = (wc[q] - mean) * rstd * ln2_g[e] + ln2_b[e];
      gate_in[(size_t)b * H4 + H2 + e] = (h16)v;
      wc_f[(size_t)b * H2 + e] = v;
    }
  }
  {
    const float* x = nt_emb2 + (size_t)nt_id[b] * H2;
    float xv[4];
#pragma unroll
    for (int q = 0; q < 4; ++q) xv[q] = x[tid + q * 256];
    float su = xv[0] + xv[1] + xv[2] + xv[3];
    float mean = block_sum256(su, s_part, tid) * (1.0f / H2);
    float sq = 0.0f;
#pragma unroll
    for (int q = 0; q < 4; ++q) { float d = xv[q] - mean; sq += d * d; }
    float var = block_sum256(sq, s_part, tid) * (1.0f / H2);
    float rstd = rsqrtf(var + 1e-5f);
#pragma unroll
    for (int q = 0; q < 4; ++q) {
      int e = tid + q * 256;
      float v = (xv[q] - mean) * rstd * ln1_g[e] + ln1_b[e];
      gate_in[(size_t)b * H4 + e] = (h16)v;
      nt2_f[(size_t)b * H2 + e] = v;
    }
  }
}

// ---------------- gate combine: g = sigmoid(lin); c = g*nt2 + (1-g)*wc ----------------
__global__ void k_gate_combine(const float* __restrict__ lin, const float* __restrict__ nt2,
                               const float* __restrict__ wc,
                               float* __restrict__ g_out, h16* __restrict__ c_h16) {
  size_t i = (size_t)blockIdx.x * blockDim.x + threadIdx.x;  // over B*H2
  if (i < (size_t)B * H2) {
    float g = sigf(lin[i]);
    g_out[i] = g;
    float c = g * nt2[i] + (1.0f - g) * wc[i];
    c_h16[i] = (h16)c;
  }
}

// ---------------- host side ----------------
static inline size_t align256(size_t x) { return (x + 255) & ~(size_t)255; }

extern "C" void kernel_launch(void* const* d_in, const int* in_sizes, int n_in,
                              void* d_out, int out_size, void* d_ws, size_t ws_size,
                              hipStream_t stream) {
  (void)in_sizes; (void)n_in; (void)out_size; (void)ws_size;

  const float* children = (const float*)d_in[0];
  const int*   ch_len   = (const int*)d_in[1];
  // d_in[2] = nt (unused by reference)
  const int*   nt_id    = (const int*)d_in[3];
  const float* stack    = (const float*)d_in[4];
  const float* Wih_f    = (const float*)d_in[5];
  const float* Whh_f    = (const float*)d_in[6];
  const float* bih_f    = (const float*)d_in[7];
  const float* bhh_f    = (const float*)d_in[8];
  const float* Wih_b    = (const float*)d_in[9];
  const float* Whh_b    = (const float*)d_in[10];
  const float* bih_b    = (const float*)d_in[11];
  const float* bhh_b    = (const float*)d_in[12];
  const float* V        = (const float*)d_in[13];
  const float* nt_emb   = (const float*)d_in[14];
  const float* nt_emb2  = (const float*)d_in[15];
  const float* ln1_g    = (const float*)d_in[16];
  const float* ln1_b    = (const float*)d_in[17];
  const float* ln2_g    = (const float*)d_in[18];
  const float* ln2_b    = (const float*)d_in[19];
  const float* gate_W   = (const float*)d_in[20];
  const float* gate_b   = (const float*)d_in[21];
  const float* out_W    = (const float*)d_in[22];
  const float* out_b    = (const float*)d_in[23];

  float* out_buf  = (float*)d_out;                         // [B, H]
  float* attn_buf = out_buf + (size_t)B * H;               // [B, L]
  float* g_buf    = attn_buf + (size_t)B * L;              // [B, 2H]

  // ---- carve workspace ----
  char* p = (char*)d_ws;
  auto carve = [&](size_t bytes) { char* r = p; p += align256(bytes); return r; };
  h16*   ch_h16   = (h16*)  carve((size_t)B * L * H * 2);
  h16*   Wihf_h   = (h16*)  carve((size_t)H4 * H * 2);
  h16*   Whhf_h   = (h16*)  carve((size_t)H4 * H * 2);
  h16*   Wihb_h   = (h16*)  carve((size_t)H4 * H * 2);
  h16*   Whhb_h   = (h16*)  carve((size_t)H4 * H * 2);
  h16*   Vt_h     = (h16*)  carve((size_t)H2 * H2 * 2);
  h16*   gW_h     = (h16*)  carve((size_t)H2 * H4 * 2);
  h16*   oW_h     = (h16*)  carve((size_t)H * H2 * 2);
  float* bias_f   = (float*)carve((size_t)H4 * 4);
  float* bias_b   = (float*)carve((size_t)H4 * 4);
  h16*   rhs_h    = (h16*)  carve((size_t)B * H2 * 2);
  float* u_f      = (float*)carve((size_t)B * H2 * 4);
  h16*   h_all    = (h16*)  carve((size_t)B * L * H2 * 2);
  h16*   h_state  = (h16*)  carve((size_t)2 * 2 * B * H * 2);  // [parity][dir][B][H]
  float* c_state  = (float*)carve((size_t)2 * B * H * 4);
  h16*   gate_in  = (h16*)  carve((size_t)B * H4 * 2);
  float* nt2_f    = (float*)carve((size_t)B * H2 * 4);
  float* wc_f     = (float*)carve((size_t)B * H2 * 4);
  float* glin_f   = (float*)carve((size_t)B * H2 * 4);
  h16*   c_h      = (h16*)  carve((size_t)B * H2 * 2);
  h16*   zerovec  = (h16*)  carve((size_t)H * 2);

  auto gblk = [](size_t n) { return (unsigned)((n + 255) / 256); };

  // ---- conversions / prep ----
  size_t nch = (size_t)B * L * H;
  k_f32_to_h16<<<gblk(nch), 256, 0, stream>>>(children, ch_h16, nch);
  k_f32_to_h16<<<gblk((size_t)H4 * H), 256, 0, stream>>>(Wih_f, Wihf_h, (size_t)H4 * H);
  k_f32_to_h16<<<gblk((size_t)H4 * H), 256, 0, stream>>>(Whh_f, Whhf_h, (size_t)H4 * H);
  k_f32_to_h16<<<gblk((size_t)H4 * H), 256, 0, stream>>>(Wih_b, Wihb_h, (size_t)H4 * H);
  k_f32_to_h16<<<gblk((size_t)H4 * H), 256, 0, stream>>>(Whh_b, Whhb_h, (size_t)H4 * H);
  k_f32_to_h16<<<gblk((size_t)H2 * H4), 256, 0, stream>>>(gate_W, gW_h, (size_t)H2 * H4);
  k_f32_to_h16<<<gblk((size_t)H * H2), 256, 0, stream>>>(out_W, oW_h, (size_t)H * H2);
  k_transpose_h16<<<gblk((size_t)H2 * H2), 256, 0, stream>>>(V, Vt_h);
  k_add2<<<gblk(H4), 256, 0, stream>>>(bih_f, bhh_f, bias_f, H4);
  k_add2<<<gblk(H4), 256, 0, stream>>>(bih_b, bhh_b, bias_b, H4);
  k_build_rhs<<<gblk((size_t)B * H2), 256, 0, stream>>>(nt_emb, nt_id, stack, rhs_h);

  // zero init LSTM state (parity-0 h, all of c) + zero vector
  k_zero_u32<<<gblk((size_t)2 * B * H / 2), 256, 0, stream>>>((unsigned*)h_state, (size_t)2 * B * H / 2);
  k_zero_u32<<<gblk((size_t)2 * B * H), 256, 0, stream>>>((unsigned*)c_state, (size_t)2 * B * H);
  k_zero_u32<<<gblk(H / 2), 256, 0, stream>>>((unsigned*)zerovec, H / 2);

  // ---- u = rhs @ V  ([B,2H] x [2H,2H]) ----
  k_wmma_gemm64<<<dim3(H2 / 64, B / 64), 32, 0, stream>>>(
      rhs_h, H2, Vt_h, H2, nullptr, u_f, H2, H2, 0);

  // ---- bidirectional LSTM: 32 serialized fused steps (GEMM + cell in one launch) ----
  const size_t hbuf = (size_t)2 * B * H;   // one parity buffer
  for (int l = 0; l < L; ++l) {
    h16* h_prev = h_state + (size_t)(l & 1) * hbuf;
    h16* h_next = h_state + (size_t)((l + 1) & 1) * hbuf;
    k_lstm_step_fused<<<dim3(H / 16, B / 256, 2), 128, 0, stream>>>(
        ch_h16, ch_len, h_prev, h_next, c_state,
        Wihf_h, Whhf_h, Wihb_h, Whhb_h, bias_f, bias_b, zerovec, h_all, l);
  }

  // ---- attention + layernorms (fused, one block per batch row) ----
  k_attn_fused<<<B, 256, 0, stream>>>(h_all, u_f, ch_len, nt_emb2, nt_id,
                                      ln1_g, ln1_b, ln2_g, ln2_b,
                                      attn_buf, gate_in, nt2_f, wc_f);

  // ---- gate GEMM: [B,4H] x [4H,2H] ----
  k_wmma_gemm64<<<dim3(H2 / 64, B / 64), 32, 0, stream>>>(
      gate_in, H4, gW_h, H4, gate_b, glin_f, H2, H4, 0);
  k_gate_combine<<<gblk((size_t)B * H2), 256, 0, stream>>>(glin_f, nt2_f, wc_f, g_buf, c_h);

  // ---- out GEMM + ReLU: [B,2H] x [2H,H] ----
  k_wmma_gemm64<<<dim3(H / 64, B / 64), 32, 0, stream>>>(
      c_h, H2, oW_h, H2, out_b, out_buf, H, H2, 1);
}
